// FIRE_DAPE_37400575214172
// MI455X (gfx1250) — matrix operand order repaired and checked by reading, not compiled, not run on hardware
//
#include <hip/hip_runtime.h>

#define S      1024
#define HEADS  12
#define BATCH  4
#define MLPW   32
#define FIRE_C 16      // fire channels padded to 16 (f16)
#define EPS    1e-6f
#define SLOPE  0.01f   // jax.nn.leaky_relu default
#define SWAPX16 0x401F // ds_swizzle: and=0x1f, or=0, xor=0x10 -> lane ^ 16

typedef __attribute__((ext_vector_type(16))) _Float16 v16h;
typedef __attribute__((ext_vector_type(8)))  _Float16 v8h;
typedef __attribute__((ext_vector_type(8)))  float    v8f;

union V16U { v16h v; int i[8]; };
union V8U  { v8h  v; int i[4]; };
union HU   { int u; _Float16 h[2]; float f; };

// A-matrix (16-bit, 16x32) element->K map: lane pair (m, m+16):
// kh=0: e0..7 -> K=e,   e8..15 -> K=e+8 (16..23)
// kh=1: e0..7 -> K=e+8, e8..15 -> K=e+16 (24..31)
__device__ __forceinline__ int a_k(int e, int kh) {
    return (e < 8 ? e : e + 8) + (kh << 3);
}

__device__ __forceinline__ float leaky(float t) {
    return fmaxf(t, SLOPE * t);   // valid since SLOPE < 1
}

// ---------------------------------------------------------------------------
// Kernel 1: fire_bias via WMMA.
//   A [16pos x 32] = leaky(nd * w1 + b1)   (rows = positions)
//   B [32 x 16]    = w2 (cols 12..15 zero)
//   D [16pos x 16ch] -> + b2 -> fire[(i*S+j)*16 + ch] f16, ch>=12 zero.
// Each wave handles 4 consecutive 16-wide j tiles.
// ---------------------------------------------------------------------------
__global__ __launch_bounds__(256) void fire_bias_kernel(
    const float* __restrict__ w1, const float* __restrict__ b1,
    const float* __restrict__ w2, const float* __restrict__ b2,
    const float* __restrict__ cp, const float* __restrict__ Lmp,
    const float* __restrict__ iLp, _Float16* __restrict__ fire)
{
    const int tid  = threadIdx.x;
    const int wave = tid >> 5;
    const int lane = tid & 31;
    const int m    = lane & 15;
    const int kh   = lane >> 4;
    const int gw   = blockIdx.x * 8 + wave;   // 16384 waves, 4 tiles each

    // per-lane A-row coefficients (hidden units this lane's K-slots cover)
    float w1v[16], b1v[16];
    #pragma unroll
    for (int e = 0; e < 16; ++e) {
        const int k = (kh << 4) + e;
        w1v[e] = w1[k];
        b1v[e] = b1[k];
    }
    // B = w2 [32 x 12pad16]; lane n holds col n, K = kh*16 + e.
    // Unconditional clamped load + select (avoids exec-masked load branches).
    const int n  = m;
    const int nc = (n < HEADS) ? n : 0;
    v16h Bw;
    #pragma unroll
    for (int e = 0; e < 16; ++e) {
        const int k = (kh << 4) + e;
        const float wv = w2[k * HEADS + nc];
        Bw[e] = (_Float16)((n < HEADS) ? wv : 0.0f);
    }
    const float b2l = b2[nc];
    const float b2v = (n < HEADS) ? b2l : 0.0f;

    const float c   = cp[0];
    const float thr = fabsf(Lmp[0] * iLp[0]);

    for (int t = 0; t < 4; ++t) {
        const int T  = gw * 4 + t;           // 16-wide tile index
        const int i  = T >> 6;               // row
        const int jb = (T & 63) << 4;        // tile j base
        const int j  = jb + m;               // this lane's position

        const float invden = 1.0f / (__logf(fabsf(c * fmaxf((float)i, thr)) + 1.0f) + EPS);
        const float rel = (j <= i) ? (float)(i - j) : 0.0f;
        const float nd  = __logf(fabsf(c * rel) + 1.0f) * invden;

        v16h Ah;
        #pragma unroll
        for (int e = 0; e < 16; ++e)
            Ah[e] = (_Float16)leaky(fmaf(nd, w1v[e], b1v[e]));

        const v8f z = {};
        v8f d = __builtin_amdgcn_wmma_f32_16x16x32_f16(
            false, Ah, false, Bw, (short)0, z, false, false);

        // D: lane n = channel, VGPR r -> pos r + 8*kh. ch 12..15 -> exactly 0.
        _Float16* dst = fire + ((size_t)i * S + jb) * FIRE_C + n;
        #pragma unroll
        for (int r = 0; r < 8; ++r) {
            const int pos = r + (kh << 3);
            dst[(size_t)pos * FIRE_C] = (_Float16)(d[r] + b2v);
        }
    }
}

// ---------------------------------------------------------------------------
// Kernel 2 (operand-swapped):
//   H^T = leaky(W3^T x X^T + b3) :  A1 = W3^T (2 tiles), B1 = X^T per j-tile
//   Y^T = W4^T x H^T + b4        :  A2 = W4^T,           B2 from half-wave swap
//   out = x + fire + Y           :  coalesced 64B row stores (lane = j)
// No LDS memory: GEMM1->GEMM2 transpose is 8x ds_swizzle SWAPX16.
// ---------------------------------------------------------------------------
__global__ __launch_bounds__(256) void fire_dape_main(
    const float* __restrict__ x, const _Float16* __restrict__ fire,
    const float* __restrict__ w3, const float* __restrict__ b3,
    const float* __restrict__ w4, const float* __restrict__ b4,
    float* __restrict__ out)
{
    const int tid  = threadIdx.x;
    const int wave = tid >> 5;
    const int lane = tid & 31;
    const int m    = lane & 15;
    const int kh   = lane >> 4;

    const int gw = blockIdx.x * 8 + wave;   // 4096 waves
    const int i  = gw >> 2;                 // row 0..1023
    const int q  = gw & 3;                  // quarter of row

    // ---- A matrices: W3^T rows hidden 0..15 / 16..31; W4^T rows = heads ----
    // All loads unconditional with clamped indices; predicate via select only.
    const int mc = (m < HEADS) ? m : 0;
    v16h A1a, A1b, A2w;
    #pragma unroll
    for (int e = 0; e < 16; ++e) {
        const int k  = (kh << 4) + e;       // K = input channel / hidden unit
        const int kc = (k < 2 * HEADS) ? k : 0;
        const float wa = w3[kc * MLPW + m];
        const float wb = w3[kc * MLPW + 16 + m];
        A1a[e] = (_Float16)((k < 2 * HEADS) ? wa : 0.0f);
        A1b[e] = (_Float16)((k < 2 * HEADS) ? wb : 0.0f);
        const float wc = w4[k * HEADS + mc];
        A2w[e] = (_Float16)((m < HEADS) ? wc : 0.0f);
    }
    // biases per D-row (row index = r + 8*kh)
    float b3lo[8], b3hi[8], b4r[8];
    #pragma unroll
    for (int r = 0; r < 8; ++r) {
        const int h  = r + (kh << 3);
        const int hc = (h < HEADS) ? h : 0;
        b3lo[r] = b3[h];
        b3hi[r] = b3[16 + h];
        const float bl = b4[hc];
        b4r[r] = (h < HEADS) ? bl : 0.0f;
    }

    const v8f z = {};

    for (int b = 0; b < BATCH; ++b) {
        for (int jt = 0; jt < 16; ++jt) {
            const int j0   = ((q << 4) + jt) << 4;
            const int jpos = j0 + m;                       // this lane's column j
            const size_t fb16 = ((size_t)i * S + jpos) * FIRE_C;

            // ---- B1 = X^T column: lane n=jpos; K = kh*16 + e ----
            V16U b1u;
            float xs[8];                                   // x ch0..7 (kh0)
            float xhi[4] = {0.f, 0.f, 0.f, 0.f};           // x ch8..11 (kh0)
            if (kh == 0) {
                #pragma unroll
                for (int c = 0; c < 8; ++c) {
                    xs[c] = x[(((size_t)b * HEADS + c) * S + i) * S + jpos];
                    b1u.v[c] = (_Float16)xs[c];
                }
                #pragma unroll
                for (int c = 0; c < 4; ++c) {
                    xhi[c] = x[(((size_t)b * HEADS + 8 + c) * S + i) * S + jpos];
                    b1u.v[8 + c] = (_Float16)xhi[c];
                }
                #pragma unroll
                for (int u = 0; u < 4; ++u)                // K12..15 = fire ch0..3
                    b1u.v[12 + u] = fire[fb16 + u];
            } else {
                #pragma unroll
                for (int u = 0; u < 8; ++u)                // K16..23 = fire ch4..11
                    b1u.v[u] = fire[fb16 + 4 + u];
                #pragma unroll
                for (int e = 8; e < 16; ++e)               // K24..31 = 0
                    b1u.v[e] = (_Float16)0.0f;
            }

            // cross-half register exchange (uniform EXEC):
            //  - kh1 lanes receive x ch8..11 (f32) for the epilogue
            //  - kh0 lanes receive fire ch4..7 (f16 pairs) for the epilogue
            HU xw0, xw1, xw2, xw3, fsw0, fsw1;
            xw0.f = xhi[0]; xw1.f = xhi[1]; xw2.f = xhi[2]; xw3.f = xhi[3];
            xw0.u = __builtin_amdgcn_ds_swizzle(xw0.u, SWAPX16);
            xw1.u = __builtin_amdgcn_ds_swizzle(xw1.u, SWAPX16);
            xw2.u = __builtin_amdgcn_ds_swizzle(xw2.u, SWAPX16);
            xw3.u = __builtin_amdgcn_ds_swizzle(xw3.u, SWAPX16);
            fsw0.u = __builtin_amdgcn_ds_swizzle(b1u.i[0], SWAPX16); // ch4,5
            fsw1.u = __builtin_amdgcn_ds_swizzle(b1u.i[1], SWAPX16); // ch6,7

            v8f d1a = __builtin_amdgcn_wmma_f32_16x16x32_f16(
                false, A1a, false, b1u.v, (short)0, z, false, false);
            v8f d1b = __builtin_amdgcn_wmma_f32_16x16x32_f16(
                false, A1b, false, b1u.v, (short)0, z, false, false);

            // ---- bias + leaky, f16-pack, half-wave swap -> B2 (H^T) ----
            V8U hau, hbu;
            #pragma unroll
            for (int r = 0; r < 8; ++r) {
                hau.v[r] = (_Float16)leaky(d1a[r] + b3lo[r]);
                hbu.v[r] = (_Float16)leaky(d1b[r] + b3hi[r]);
            }
            int sa[4], sb[4];
            #pragma unroll
            for (int p = 0; p < 4; ++p) {
                sa[p] = __builtin_amdgcn_ds_swizzle(hau.i[p], SWAPX16);
                sb[p] = __builtin_amdgcn_ds_swizzle(hbu.i[p], SWAPX16);
            }
            // kh0: K0..15  = own d1a (h0..7)      | partner d1a (h8..15)
            // kh1: K16..31 = partner d1b (h16..23)| own d1b (h24..31)
            V16U b2u;
            #pragma unroll
            for (int p = 0; p < 4; ++p) {
                b2u.i[p]     = kh ? sb[p] : hau.i[p];
                b2u.i[4 + p] = kh ? hbu.i[p] : sa[p];
            }

            v8f d2 = __builtin_amdgcn_wmma_f32_16x16x32_f16(
                false, A2w, false, b2u.v, (short)0, z, false, false);

            // ---- epilogue: out = x + fire + (y + b4); lane = j (coalesced) ----
            if (kh == 0) {
                #pragma unroll
                for (int r = 0; r < 8; ++r) {              // heads 0..7
                    float fv;
                    if (r < 4)      fv = (float)b1u.v[12 + r];            // ch0..3
                    else if (r < 6) fv = (float)fsw0.h[r - 4];            // ch4,5
                    else            fv = (float)fsw1.h[r - 6];            // ch6,7
                    const size_t oi = (((size_t)b * HEADS + r) * S + i) * S + jpos;
                    out[oi] = xs[r] + fv + (d2[r] + b4r[r]);
                }
            } else {
                const float xv[4] = {xw0.f, xw1.f, xw2.f, xw3.f};
                #pragma unroll
                for (int r = 0; r < 4; ++r) {              // heads 8..11
                    const float fv = (float)b1u.v[4 + r];                 // ch8..11
                    const size_t oi = (((size_t)b * HEADS + 8 + r) * S + i) * S + jpos;
                    out[oi] = xv[r] + fv + (d2[r] + b4r[r]);
                }
            }
        }
    }
}

extern "C" void kernel_launch(void* const* d_in, const int* in_sizes, int n_in,
                              void* d_out, int out_size, void* d_ws, size_t ws_size,
                              hipStream_t stream) {
    // 0:x 1:c 2:L_multiplier 3:init_L 4:w1 5:b1 6:w2 7:b2 8:w3 9:b3 10:w4 11:b4
    const float* x  = (const float*)d_in[0];
    const float* cp = (const float*)d_in[1];
    const float* Lm = (const float*)d_in[2];
    const float* iL = (const float*)d_in[3];
    const float* w1 = (const float*)d_in[4];
    const float* b1 = (const float*)d_in[5];
    const float* w2 = (const float*)d_in[6];
    const float* b2 = (const float*)d_in[7];
    const float* w3 = (const float*)d_in[8];
    const float* b3 = (const float*)d_in[9];
    const float* w4 = (const float*)d_in[10];
    const float* b4 = (const float*)d_in[11];
    float* out = (float*)d_out;

    _Float16* fire = (_Float16*)d_ws;   // S*S*16 halves = 32 MB scratch

    // Phase A: fire_bias via WMMA. 16384 waves x 4 tiles -> 2048 blocks.
    fire_bias_kernel<<<(S * S) / (64 * 8), 256, 0, stream>>>(
        w1, b1, w2, b2, cp, Lm, iL, fire);

    // Phase B: fused MLP2 + residual. 4096 waves (row-quarters) -> 512 blocks.
    fire_dape_main<<<(S * 4) / 8, 256, 0, stream>>>(x, fire, w3, b3, w4, b4, out);
}
